// QKVAttention_66511863546393
// MI455X (gfx1250) — matrix operand order repaired and checked by reading, not compiled, not run on hardware
//
#include <hip/hip_runtime.h>

typedef __attribute__((ext_vector_type(16))) __bf16 v16bf;
typedef __attribute__((ext_vector_type(8)))  float  v8f;
typedef __attribute__((ext_vector_type(8)))  unsigned short v8us;

#define N_SEQ   2048
#define D_HEAD  64
#define TILE_I  128            // queries per workgroup (8 waves x 16)
#define TILE_J  64             // keys per inner tile
#define NTILES  (N_SEQ / TILE_J)
#define NWAVES  8
// fold 1/sqrt(d) (= both d^-1/4 factors) and log2(e) so softmax runs in exp2 domain
#define Q_SCALE (0.125f * 1.4426950408889634f)

// LDS pool offsets (ushort units)
#define OFF_Q   0        // Qs  : bf16 [128][64]  (16 KB)
#define OFF_K   8192     // Ks0 : bf16 [64][64]   (8 KB)  buffer b at OFF_K + b*8192
#define OFF_V   12288    // Vs0 : bf16 [64][64]   (8 KB)  buffer b at OFF_V + b*8192
#define OFF_P   24576    // Ps  : bf16 [8][16][64] (16 KB)
#define BUF_STRIDE 8192  // K/V double-buffer stride (ushorts)

__device__ __forceinline__ unsigned short f2bf(float f) {
  union { float f; unsigned u; } v; v.f = f;
  unsigned u = v.u;
  unsigned r = 0x7FFFu + ((u >> 16) & 1u);   // round-to-nearest-even
  return (unsigned short)((u + r) >> 16);
}

__device__ __forceinline__ unsigned pack2(float lo, float hi) {
  return (unsigned)f2bf(lo) | ((unsigned)f2bf(hi) << 16);
}

// Assemble a 16-element bf16 fragment from two 16-byte LDS chunks.
__device__ __forceinline__ v16bf frag2(const unsigned short* p0, const unsigned short* p1) {
  union { struct { v8us lo; v8us hi; } s; v16bf v; } u;
  u.s.lo = *(const v8us*)p0;
  u.s.hi = *(const v8us*)p1;
  return u.v;
}

__device__ __forceinline__ v8f zero8() {
  v8f z;
#pragma unroll
  for (int i = 0; i < 8; ++i) z[i] = 0.0f;
  return z;
}

__global__ __launch_bounds__(256)
void qkv_attention_wmma_kernel(const float* __restrict__ qkv, float* __restrict__ out) {
  // 64 KB LDS pool; first 32 KB reused as float Os[64][128] after the main loop.
  __shared__ __attribute__((aligned(16))) unsigned short smem[32768];
  unsigned short* Qs = smem + OFF_Q;
  unsigned short* Ps = smem + OFF_P;

  const int tid  = threadIdx.x;
  const int wave = tid >> 5;
  const int lane = tid & 31;
  const int half = lane >> 4;   // 16-lane half of the wave
  const int col  = lane & 15;   // N index in a 16-wide tile (also A-fragment row)

  const int head = blockIdx.x >> 4;   // 0..31  (b * 16 + h)
  const int tile = blockIdx.x & 15;   // query tile 0..15
  const int bb   = head >> 4;
  const int h    = head & 15;
  const int i0   = tile * TILE_I;

  const size_t plane = (size_t)N_SEQ;
  const float* qb = qkv + ((size_t)bb * 3072 +        (size_t)h * 64) * plane;
  const float* kb = qkv + ((size_t)bb * 3072 + 1024 + (size_t)h * 64) * plane;
  const float* vb = qkv + ((size_t)bb * 3072 + 2048 + (size_t)h * 64) * plane;

  // ---- Stage Q (scaled), transposed to [i][d]
  for (int idx = tid; idx < TILE_I * D_HEAD; idx += 256) {
    int dd = idx >> 7;
    int ii = idx & 127;          // contiguous across tid -> coalesced
    Qs[ii * D_HEAD + dd] = f2bf(qb[(size_t)dd * plane + i0 + ii] * Q_SCALE);
  }
  // ---- Prologue: stage K/V tile 0 into buffer 0
  {
    unsigned short* Kn = smem + OFF_K;
    unsigned short* Vn = smem + OFF_V;
#pragma unroll
    for (int p = 0; p < 8; ++p) {
      int idx2 = tid + p * 256;           // 0..2047 (pairs)
      int dd = idx2 >> 5;
      int jj = (idx2 & 31) << 1;
      float2 kk2 = *(const float2*)(kb + (size_t)dd * plane + jj);
      float2 vv2 = *(const float2*)(vb + (size_t)dd * plane + jj);
      Kn[jj * D_HEAD + dd]       = f2bf(kk2.x);
      Kn[(jj + 1) * D_HEAD + dd] = f2bf(kk2.y);
      *(unsigned*)&Vn[dd * TILE_J + jj] = pack2(vv2.x, vv2.y);
    }
  }
  __syncthreads();

  v8f oacc[4];
  float m_run[8], l_run[8];
#pragma unroll
  for (int t = 0; t < 4; ++t) oacc[t] = zero8();
#pragma unroll
  for (int v = 0; v < 8; ++v) { m_run[v] = -1e30f; l_run[v] = 0.0f; }

  unsigned short* Pw = Ps + wave * 16 * TILE_J;
  const int qrow_base = (wave * 16 + col) * D_HEAD;

  for (int kt = 0; kt < NTILES; ++kt) {
    const int cur = kt & 1, nxt = cur ^ 1;

    // ---- Issue next tile's global loads into registers (latency overlapped with compute)
    float kx[16], vx[16];
    if (kt + 1 < NTILES) {
      const int jn = (kt + 1) * TILE_J;
#pragma unroll
      for (int p = 0; p < 8; ++p) {
        int idx2 = tid + p * 256;
        int dd = idx2 >> 5;
        int jj = (idx2 & 31) << 1;
        float2 kk2 = *(const float2*)(kb + (size_t)dd * plane + jn + jj);
        float2 vv2 = *(const float2*)(vb + (size_t)dd * plane + jn + jj);
        kx[2 * p] = kk2.x; kx[2 * p + 1] = kk2.y;
        vx[2 * p] = vv2.x; vx[2 * p + 1] = vv2.y;
      }
    }
    if (kt + 2 < NTILES) {   // deep prefetch (global_prefetch_b8)
      int dd = tid >> 2;
      int jq = (tid & 3) * 16;
      __builtin_prefetch(&kb[(size_t)dd * plane + (kt + 2) * TILE_J + jq], 0, 0);
      __builtin_prefetch(&vb[(size_t)dd * plane + (kt + 2) * TILE_J + jq], 0, 0);
    }

    // ---- S = Q^T K : 16 queries x 64 keys per wave (8 WMMAs)
    const unsigned short* Kc = smem + OFF_K + cur * BUF_STRIDE;
    const unsigned short* Vc = smem + OFF_V + cur * BUF_STRIDE;
    v8f sacc[4];
#pragma unroll
    for (int t = 0; t < 4; ++t) {
      v8f c = zero8();
#pragma unroll
      for (int kk = 0; kk < D_HEAD; kk += 32) {
        const unsigned short* qp = Qs + qrow_base + kk;
        v16bf a = frag2(qp + half * 8, qp + 16 + half * 8);
        const unsigned short* kp = Kc + (t * 16 + col) * D_HEAD + kk + half * 8;
        v16bf b = frag2(kp, kp + 16);
        c = __builtin_amdgcn_wmma_f32_16x16x32_bf16(false, a, false, b,
                                                    (short)0, c, false, false);
      }
      sacc[t] = c;
    }

    // ---- Online softmax in exp2 domain. Row M = v + 8*half spans a 16-lane half.
#pragma unroll
    for (int v = 0; v < 8; ++v) {
      float mx = fmaxf(fmaxf(sacc[0][v], sacc[1][v]), fmaxf(sacc[2][v], sacc[3][v]));
#pragma unroll
      for (int s = 1; s < 16; s <<= 1) mx = fmaxf(mx, __shfl_xor(mx, s, 16));
      float mnew = fmaxf(m_run[v], mx);
      float corr = exp2f(m_run[v] - mnew);
      float rs = 0.0f;
#pragma unroll
      for (int t = 0; t < 4; ++t) {
        float p = exp2f(sacc[t][v] - mnew);
        sacc[t][v] = p;
        rs += p;
      }
#pragma unroll
      for (int s = 1; s < 16; s <<= 1) rs += __shfl_xor(rs, s, 16);
      l_run[v] = l_run[v] * corr + rs;
      m_run[v] = mnew;
#pragma unroll
      for (int t = 0; t < 4; ++t) oacc[t][v] *= corr;
      int prow = v + 8 * half;   // C layout -> row-major [i][j] in per-wave LDS patch
#pragma unroll
      for (int t = 0; t < 4; ++t)
        Pw[prow * TILE_J + t * 16 + col] = f2bf(sacc[t][v]);
    }

    // ---- O += P * V^T : 16 queries x 64 features (8 WMMAs)
#pragma unroll
    for (int t = 0; t < 4; ++t) {
      v8f c = oacc[t];
#pragma unroll
      for (int kk = 0; kk < TILE_J; kk += 32) {
        const unsigned short* pp = Pw + col * TILE_J + kk;
        v16bf a = frag2(pp + half * 8, pp + 16 + half * 8);
        const unsigned short* vp = Vc + (t * 16 + col) * TILE_J + kk + half * 8;
        v16bf b = frag2(vp, vp + 16);
        c = __builtin_amdgcn_wmma_f32_16x16x32_bf16(false, a, false, b,
                                                    (short)0, c, false, false);
      }
      oacc[t] = c;
    }

    // ---- Convert + store the prefetched tile into the other buffer
    // (buf[nxt] was last read before the previous barrier -> safe to overwrite)
    if (kt + 1 < NTILES) {
      unsigned short* Kn = smem + OFF_K + nxt * BUF_STRIDE;
      unsigned short* Vn = smem + OFF_V + nxt * BUF_STRIDE;
#pragma unroll
      for (int p = 0; p < 8; ++p) {
        int idx2 = tid + p * 256;
        int dd = idx2 >> 5;
        int jj = (idx2 & 31) << 1;
        Kn[jj * D_HEAD + dd]       = f2bf(kx[2 * p]);
        Kn[(jj + 1) * D_HEAD + dd] = f2bf(kx[2 * p + 1]);
        *(unsigned*)&Vn[dd * TILE_J + jj] = pack2(vx[2 * p], vx[2 * p + 1]);
      }
    }
    __syncthreads();
  }

  // ---- Transpose O through LDS for coalesced output stores
  float* Os = (float*)smem;                 // [64][128] fp32, reuses first 32 KB
  __syncthreads();                          // all waves done reading Qs/K/V buffers
#pragma unroll
  for (int v = 0; v < 8; ++v) {
    float rl = 1.0f / l_run[v];
    int il = wave * 16 + v + 8 * half;
#pragma unroll
    for (int t = 0; t < 4; ++t)
      Os[(t * 16 + col) * TILE_I + il] = oacc[t][v] * rl;
  }
  __syncthreads();

  float* ob = out + ((size_t)bb * 1024 + (size_t)h * 64) * plane;
  for (int idx = tid; idx < D_HEAD * TILE_I; idx += 256) {
    int dd = idx >> 7;
    int ii = idx & 127;                     // contiguous -> 128B/wave coalesced stores
    ob[(size_t)dd * plane + i0 + ii] = Os[idx];
  }
}

extern "C" void kernel_launch(void* const* d_in, const int* in_sizes, int n_in,
                              void* d_out, int out_size, void* d_ws, size_t ws_size,
                              hipStream_t stream) {
  (void)in_sizes; (void)n_in; (void)out_size; (void)d_ws; (void)ws_size;
  const float* qkv = (const float*)d_in[0];
  float* out = (float*)d_out;
  dim3 grid(32 * (N_SEQ / TILE_I));   // 32 head-batches x 16 query tiles = 512 WGs
  dim3 block(256);                    // 8 wave32 per WG
  hipLaunchKernelGGL(qkv_attention_wmma_kernel, grid, block, 0, stream, qkv, out);
}
